// LSTMModel_6914897347035
// MI455X (gfx1250) — matrix operand order repaired and checked by reading, not compiled, not run on hardware
//
#include <hip/hip_runtime.h>

typedef __attribute__((ext_vector_type(16))) _Float16 v16h;
typedef __attribute__((ext_vector_type(8)))  _Float16 v8h;
typedef __attribute__((ext_vector_type(8)))  float    v8f;
typedef __attribute__((ext_vector_type(2)))  float    v2f;

#define LSTM_IN   3
#define LSTM_H    32
#define LSTM_T    512
#define LSTM_B    8192
#define ROW_PITCH 40   // halfs per LDS row (80 B pitch -> 16B aligned, conflict-free b128)

#ifndef __has_builtin
#define __has_builtin(x) 0
#endif

#if __has_builtin(__builtin_amdgcn_tanhf)
#define FAST_TANH(v) __builtin_amdgcn_tanhf(v)           // v_tanh_f32 (gfx1250 TRANS op)
#else
#define FAST_TANH(v) (1.0f - 2.0f * __builtin_amdgcn_rcpf(1.0f + __builtin_amdgcn_exp2f(2.8853900817779268f * (v))))
#endif

// sigmoid(x) = 0.5 + 0.5 * tanh(x/2)  -> single TRANS op
__device__ __forceinline__ float fast_sigmoid(float x) {
  return 0.5f + 0.5f * FAST_TANH(0.5f * x);
}

__launch_bounds__(64)
__global__ void lstm_wmma_kernel(const float* __restrict__ x,
                                 const float* __restrict__ W_ih,
                                 const float* __restrict__ W_hh,
                                 const float* __restrict__ b_ih,
                                 const float* __restrict__ b_hh,
                                 const float* __restrict__ W_fc,
                                 const float* __restrict__ b_fc,
                                 float* __restrict__ out) {
  // Double-buffered h tile (f16, row-major [16][32] with pad): one barrier/step.
  __shared__ __align__(16) _Float16 h_a[16 * ROW_PITCH];
  __shared__ __align__(16) _Float16 h_b[16 * ROW_PITCH];

  const int tid  = threadIdx.x;
  const int wave = tid >> 5;          // 0/1: which 16-wide gate-column half this wave owns
  const int lane = tid & 31;
  const int l15  = lane & 15;
  const int hi   = lane >> 4;         // K-half (A/B operands) and M-half (C/D) selector
  const int row0 = blockIdx.x * 16;   // batch rows of this block

  // ---- Stationary operands, built once, resident in VGPRs ------------------
  // Gate chunks G: 0=i, 1=f, 2=g, 3=o. This wave's columns: tile n = wave + 2G,
  // gate index g = n*16 + l15.
  v16h bh[4];   // W_hh^T fragments (16-bit B 32x16, this code's contiguous k order: lane<16 K=0..15, lane>=16 K=16..31)
  v2f  bx[4];   // f32 B 4x16: [W_ih^T (K=0..2); b_ih+b_hh (K=3)]; lane<16 K=0,1; lane>=16 K=2,3
  #pragma unroll
  for (int G = 0; G < 4; ++G) {
    const int g = (wave + 2 * G) * 16 + l15;
    const float* wrow = W_hh + g * LSTM_H + 16 * hi;
    #pragma unroll
    for (int e = 0; e < 16; ++e) bh[G][e] = (_Float16)wrow[e];
    if (hi == 0) { bx[G].x = W_ih[g * LSTM_IN + 0]; bx[G].y = W_ih[g * LSTM_IN + 1]; }
    else         { bx[G].x = W_ih[g * LSTM_IN + 2]; bx[G].y = b_ih[g] + b_hh[g];     }
  }

  // ---- h_{-1} = 0 (only the first-read buffer needs init) ----
  for (int i = tid; i < 16 * ROW_PITCH; i += 64) h_a[i] = (_Float16)0.0f;
  __syncthreads();

  // c state: element r <-> (m = r + 8*hi, j = wave*16 + l15)  [C/D f32 layout]
  float c[8];
  #pragma unroll
  for (int r = 0; r < 8; ++r) c[r] = 0.0f;

  const float* xrow = x + (size_t)(row0 + l15) * LSTM_T * LSTM_IN;

  // One LSTM timestep: read h_{t-1} from src, write h_t to dst, one barrier.
  auto step = [&](const _Float16* __restrict__ src, _Float16* __restrict__ dst, int t) {
    const float* xp = xrow + t * LSTM_IN;

    // A_x (f32 16x4 layout): lane<16 -> M=l15, K0=x0,K1=x1 ; lane>=16 -> K2=x2,K3=1
    v2f ax;
    if (hi == 0) {
      ax.x = xp[0]; ax.y = xp[1];
      if (t + 24 < LSTM_T) __builtin_prefetch(xp + 24 * LSTM_IN, 0, 0);
    } else {
      ax.x = xp[2]; ax.y = 1.0f;
    }

    // A_h (f16 16x32, this code's contiguous k order, same as bh): lane<16 -> M=l15, K=0..15; lane>=16 -> K=16..31
    v8h alo = *(const v8h*)&src[l15 * ROW_PITCH + 16 * hi];       // ds_load_b128
    v8h ahi = *(const v8h*)&src[l15 * ROW_PITCH + 16 * hi + 8];   // ds_load_b128
    v16h ah;
    #pragma unroll
    for (int e = 0; e < 8; ++e) { ah[e] = alo[e]; ah[8 + e] = ahi[e]; }

    // gates = [x,1] @ [W_ih^T; b]  (fp32 WMMA)  +  h @ W_hh^T  (f16 WMMA, f32 acc)
    v8f z = {};
    v8f acc_i = __builtin_amdgcn_wmma_f32_16x16x4_f32(false, ax, false, bx[0], (short)0, z, false, false);
    v8f acc_f = __builtin_amdgcn_wmma_f32_16x16x4_f32(false, ax, false, bx[1], (short)0, z, false, false);
    v8f acc_g = __builtin_amdgcn_wmma_f32_16x16x4_f32(false, ax, false, bx[2], (short)0, z, false, false);
    v8f acc_o = __builtin_amdgcn_wmma_f32_16x16x4_f32(false, ax, false, bx[3], (short)0, z, false, false);
    acc_i = __builtin_amdgcn_wmma_f32_16x16x32_f16(false, ah, false, bh[0], (short)0, acc_i, false, false);
    acc_f = __builtin_amdgcn_wmma_f32_16x16x32_f16(false, ah, false, bh[1], (short)0, acc_f, false, false);
    acc_g = __builtin_amdgcn_wmma_f32_16x16x32_f16(false, ah, false, bh[2], (short)0, acc_g, false, false);
    acc_o = __builtin_amdgcn_wmma_f32_16x16x32_f16(false, ah, false, bh[3], (short)0, acc_o, false, false);

    #pragma unroll
    for (int r = 0; r < 8; ++r) {
      const float gi = fast_sigmoid(acc_i[r]);
      const float gf = fast_sigmoid(acc_f[r]);
      const float gg = FAST_TANH(acc_g[r]);
      const float go = fast_sigmoid(acc_o[r]);
      const float cn = gf * c[r] + gi * gg;
      c[r] = cn;
      const float hn = go * FAST_TANH(cn);
      dst[(r + 8 * hi) * ROW_PITCH + wave * 16 + l15] = (_Float16)hn;  // ds_store_b16
    }
    __syncthreads();   // h_t visible; next step writes the other buffer (no WAR)
  };

  for (int t = 0; t < LSTM_T; t += 2) {
    step(h_a, h_b, t);
    step(h_b, h_a, t + 1);
  }

  // ---- out[b] = h_T[b] @ W_fc^T + b_fc  (OUT=2, trivial); h_T is in h_a ----
  if (tid < 16) {
    float a0 = b_fc[0], a1 = b_fc[1];
    #pragma unroll
    for (int k = 0; k < LSTM_H; ++k) {
      const float hv = (float)h_a[tid * ROW_PITCH + k];
      a0 += hv * W_fc[k];
      a1 += hv * W_fc[LSTM_H + k];
    }
    out[(row0 + tid) * 2 + 0] = a0;
    out[(row0 + tid) * 2 + 1] = a1;
  }
}

extern "C" void kernel_launch(void* const* d_in, const int* in_sizes, int n_in,
                              void* d_out, int out_size, void* d_ws, size_t ws_size,
                              hipStream_t stream) {
  (void)in_sizes; (void)n_in; (void)d_ws; (void)ws_size; (void)out_size;
  const float* x    = (const float*)d_in[0];
  const float* W_ih = (const float*)d_in[1];
  const float* W_hh = (const float*)d_in[2];
  const float* b_ih = (const float*)d_in[3];
  const float* b_hh = (const float*)d_in[4];
  const float* W_fc = (const float*)d_in[5];
  const float* b_fc = (const float*)d_in[6];
  float* out = (float*)d_out;

  dim3 grid(LSTM_B / 16);   // 512 blocks
  dim3 block(64);           // 2 wave32 waves per block
  lstm_wmma_kernel<<<grid, block, 0, stream>>>(x, W_ih, W_hh, b_ih, b_hh, W_fc, b_fc, out);
}
